// LeftHemisphereV2_18769007084195
// MI455X (gfx1250) — matrix operand order repaired and checked by reading, not compiled, not run on hardware
//
#include <hip/hip_runtime.h>
#include <hip/hip_bf16.h>
#include <math.h>

// ---------------------------------------------------------------------------
// CDNA5 (gfx1250) implementation: bf16 WMMA GEMMs + fused LSTM + flash attn.
// Dims from the reference: B=32, L=512 (captions[:, :-1]), H=384, E=192,
// X=E+H=576, 4H=1536, V=5000, NH=4, HD=96, GH=192.
// ---------------------------------------------------------------------------

typedef __bf16 bf16_t;
typedef __attribute__((ext_vector_type(8)))  __bf16 v8bf;
typedef __attribute__((ext_vector_type(16))) __bf16 v16bf;
typedef __attribute__((ext_vector_type(8)))  float  v8f;

#define WMMA_BF16(a, b, c) \
  __builtin_amdgcn_wmma_f32_16x16x32_bf16(false, (a), false, (b), (short)0, (c), false, false)

constexpr int BB  = 32;
constexpr int LL  = 512;
constexpr int HH  = 384;
constexpr int EE  = 192;
constexpr int XX  = 576;    // E + H
constexpr int G4  = 1536;   // 4*H
constexpr int VV  = 5000;
constexpr int NHD = 4;
constexpr int HD  = 96;
constexpr int GHH = 192;

__device__ __forceinline__ float sigm(float x) { return 1.0f / (1.0f + __expf(-x)); }

// ---------------------------------------------------------------------------
// Fragment loaders (wave32; contiguous k order, same for A and B, not the ISA table order).
// A (16x32 bf16): lane = M (lane&15); half=lane>>4 holds K = k0+16h .. k0+16h+15.
// B (32x16 bf16): lane = N (lane&15); half holds K = k0+16h .. k0+16h+15.
// ---------------------------------------------------------------------------
__device__ __forceinline__ v16bf frag_a_bf16(const bf16_t* A, int lda, int m0, int k0, int lane) {
  int m = m0 + (lane & 15), half = lane >> 4;
  const bf16_t* row = A + (size_t)m * lda + k0 + half * 16;
  v8bf lo = *(const v8bf*)(row);
  v8bf hi = *(const v8bf*)(row + 8);
  return __builtin_shufflevector(lo, hi, 0, 1, 2, 3, 4, 5, 6, 7, 8, 9, 10, 11, 12, 13, 14, 15);
}

__device__ __forceinline__ v16bf frag_b_rowmajor(const bf16_t* W, int ldw, int n0, int k0, int N, int lane) {
  int n = n0 + (lane & 15), half = lane >> 4;
  if (n < N) {
    return *(const v16bf*)(W + (size_t)n * ldw + k0 + half * 16);
  }
  v16bf z;
#pragma unroll
  for (int i = 0; i < 16; i++) z[i] = (__bf16)0.0f;
  return z;
}

// ---------------------------------------------------------------------------
// Utility kernels
// ---------------------------------------------------------------------------
__global__ void k_f32_to_bf16(const float* __restrict__ s, bf16_t* __restrict__ d, size_t n) {
  size_t i  = blockIdx.x * (size_t)blockDim.x + threadIdx.x;
  size_t st = (size_t)gridDim.x * blockDim.x;
  for (; i < n; i += st) d[i] = (__bf16)s[i];
}

// x = [emb[captions[b, l]], visual_context[b]]  -> bf16 (B*L, 576)
__global__ void k_build_x(const int* __restrict__ cap, const float* __restrict__ emb,
                          const float* __restrict__ vis, bf16_t* __restrict__ x) {
  size_t n = (size_t)BB * LL * XX;
  for (size_t i = blockIdx.x * (size_t)blockDim.x + threadIdx.x; i < n;
       i += (size_t)gridDim.x * blockDim.x) {
    size_t row = i / XX;
    int col = (int)(i % XX);
    int b = (int)(row / LL), l = (int)(row % LL);
    float v;
    if (col < EE) {
      int tok = cap[b * (LL + 1) + l];   // captions is (B, 513); take [:, :-1]
      v = emb[(size_t)tok * EE + col];
    } else {
      v = vis[b * HH + (col - EE)];
    }
    x[i] = (__bf16)v;
  }
}

// ---------------------------------------------------------------------------
// Generic GEMM: C[M,N] = A[M,K] @ W[N,K]^T + bias1 + bias2
// Workgroup = 8 waves; each wave owns a 16x32 C strip (two tiles sharing the
// A fragment -> 2 WMMAs per A load); block tile 16x256.
// ---------------------------------------------------------------------------
template <bool OUT_BF16>
__global__ __launch_bounds__(256) void k_gemm(const bf16_t* __restrict__ A,
                                              const bf16_t* __restrict__ W,
                                              const float* __restrict__ b1,
                                              const float* __restrict__ b2,
                                              void* __restrict__ Cout, int M, int N, int K) {
  int lane = threadIdx.x & 31, wave = threadIdx.x >> 5;
  int n0 = blockIdx.x * 256 + wave * 32;
  int m0 = blockIdx.y * 16;
  if (n0 >= N) return;  // wave-uniform

  v8f acc0 = {}, acc1 = {};
  for (int k0 = 0; k0 < K; k0 += 32) {
    v16bf a  = frag_a_bf16(A, K, m0, k0, lane);
    v16bf b0 = frag_b_rowmajor(W, K, n0, k0, N, lane);
    v16bf b1f = frag_b_rowmajor(W, K, n0 + 16, k0, N, lane);
    acc0 = WMMA_BF16(a, b0, acc0);
    acc1 = WMMA_BF16(a, b1f, acc1);
  }
  int half = lane >> 4, lc = lane & 15;
#pragma unroll
  for (int tt = 0; tt < 2; tt++) {
    int n = n0 + tt * 16 + lc;
    v8f acc = tt ? acc1 : acc0;
    if (n < N) {
      float bias = (b1 ? b1[n] : 0.0f) + (b2 ? b2[n] : 0.0f);
#pragma unroll
      for (int r = 0; r < 8; r++) {
        int m = m0 + r + 8 * half;
        float v = acc[r] + bias;
        if (OUT_BF16)
          ((bf16_t*)Cout)[(size_t)m * N + n] = (__bf16)v;
        else
          ((float*)Cout)[(size_t)m * N + n] = v;
      }
    }
  }
}

// ---------------------------------------------------------------------------
// LSTM recurrence: one workgroup (768 threads = 24 waves), 512 sequential
// steps. g = pre[:,t,:] + h @ Whh^T, gates -> (h, c). h kept in LDS (bf16),
// c in registers. Each wave owns 2 units processed SEQUENTIALLY (a unit =
// (m-tile, j-tile), computing all four gate tiles) so only 4 v8f accumulators
// are live at a time. K-loop is kept rolled so Whh fragments stream from L2
// each step instead of being LICM-hoisted and spilled to scratch.
// ---------------------------------------------------------------------------
__global__ __launch_bounds__(768) void k_lstm(const float* __restrict__ pre,
                                              const bf16_t* __restrict__ Whh,
                                              const float* __restrict__ vis,
                                              float* __restrict__ hout,
                                              bf16_t* __restrict__ hout_bf) {
  __shared__ bf16_t h_bf[BB * HH];  // 24 KB
  int tid = threadIdx.x, lane = tid & 31, wave = tid >> 5;  // wave: 0..23
  int half = lane >> 4, lc = lane & 15;

  for (int i = tid; i < BB * HH; i += 768) h_bf[i] = (__bf16)vis[i];
  __syncthreads();

  float creg[16], hreg[16];
#pragma unroll
  for (int i = 0; i < 16; i++) creg[i] = 0.0f;

  for (int t = 0; t < LL; t++) {
#pragma unroll 1
    for (int s = 0; s < 2; s++) {
      int u = wave * 2 + s;
      int mt = u / 24, jt = u % 24;
      int col = jt * 16 + lc;
      v8f acc0 = {}, acc1 = {}, acc2 = {}, acc3 = {};
#pragma clang loop unroll(disable)
      for (int k0 = 0; k0 < HH; k0 += 32) {
        // A fragment from LDS-resident h (two ds_load_b128)
        int m = mt * 16 + lc;
        const bf16_t* row = h_bf + m * HH + k0 + half * 16;
        v8bf lo = *(const v8bf*)(row);
        v8bf hi = *(const v8bf*)(row + 8);
        v16bf a = __builtin_shufflevector(lo, hi, 0, 1, 2, 3, 4, 5, 6, 7,
                                          8, 9, 10, 11, 12, 13, 14, 15);
        const bf16_t* wbase = Whh + (size_t)(jt * 16 + lc) * HH + k0 + half * 16;
        const size_t gstep = (size_t)HH * HH;  // 384 rows * 384
        acc0 = WMMA_BF16(a, *(const v16bf*)(wbase + 0 * gstep), acc0);
        acc1 = WMMA_BF16(a, *(const v16bf*)(wbase + 1 * gstep), acc1);
        acc2 = WMMA_BF16(a, *(const v16bf*)(wbase + 2 * gstep), acc2);
        acc3 = WMMA_BF16(a, *(const v16bf*)(wbase + 3 * gstep), acc3);
      }
      // Gate math straight out of the accumulators; defer LDS writes.
#pragma unroll
      for (int r = 0; r < 8; r++) {
        int b = mt * 16 + r + 8 * half;  // batch index
        size_t prow = ((size_t)b * LL + t) * G4;
        float gi = acc0[r] + pre[prow + 0 * HH + col];
        float gf = acc1[r] + pre[prow + 1 * HH + col];
        float gg = acc2[r] + pre[prow + 2 * HH + col];
        float go = acc3[r] + pre[prow + 3 * HH + col];
        float c = creg[s * 8 + r];
        c = sigm(gf) * c + sigm(gi) * tanhf(gg);
        creg[s * 8 + r] = c;
        hreg[s * 8 + r] = sigm(go) * tanhf(c);
      }
    }
    __syncthreads();  // everyone done reading h_bf for step t
#pragma unroll
    for (int s = 0; s < 2; s++) {
      int u = wave * 2 + s;
      int mt = u / 24, jt = u % 24;
      int col = jt * 16 + lc;
#pragma unroll
      for (int r = 0; r < 8; r++) {
        int b = mt * 16 + r + 8 * half;
        float h = hreg[s * 8 + r];
        size_t orow = ((size_t)b * LL + t) * HH + col;
        hout[orow] = h;
        hout_bf[orow] = (__bf16)h;
        h_bf[b * HH + col] = (__bf16)h;
      }
    }
    __syncthreads();  // h_bf updated for step t+1
  }
}

// ---------------------------------------------------------------------------
// Fused attention, flash style: one wave per (batch, head, q-tile of 16).
// Heads 0,1: causal + local_scale; heads 2,3: full + global_scale.
// qkv is bf16 (B*L, 1152) with column = (which*NH + h)*HD + d.
// ---------------------------------------------------------------------------
__global__ __launch_bounds__(32) void k_attn(const bf16_t* __restrict__ qkv,
                                             const float* __restrict__ lsc,
                                             const float* __restrict__ gsc,
                                             bf16_t* __restrict__ ctx) {
  __shared__ float Pld[16 * 32];  // P tile staging: C-layout -> A-layout
  int lane = threadIdx.x, lc = lane & 15, half = lane >> 4;
  int qt = blockIdx.x, h = blockIdx.y, b = blockIdx.z;
  bool causal = (h < 2);
  float scale = (causal ? lsc[0] : gsc[0]) * 0.1020620726159658f;  // 1/sqrt(96)

  const bf16_t* Qb = qkv + (size_t)b * LL * (3 * HH) + (0 * NHD + h) * HD;
  const bf16_t* Kb = qkv + (size_t)b * LL * (3 * HH) + (1 * NHD + h) * HD;
  const bf16_t* Vb = qkv + (size_t)b * LL * (3 * HH) + (2 * NHD + h) * HD;

  v16bf qa[3];
#pragma unroll
  for (int dk = 0; dk < 3; dk++) qa[dk] = frag_a_bf16(Qb, 3 * HH, qt * 16, dk * 32, lane);

  v8f o[6] = {};
  float rm[8], rs[8];
#pragma unroll
  for (int r = 0; r < 8; r++) { rm[r] = -3.0e38f; rs[r] = 0.0f; }

  int jend = causal ? (qt * 16 + 16) : LL;
  for (int j0 = 0; j0 < jend; j0 += 32) {
    v8f s0 = {}, s1 = {};
#pragma unroll
    for (int dk = 0; dk < 3; dk++) {
      v16bf k0f = frag_b_rowmajor(Kb, 3 * HH, j0, dk * 32, LL, lane);
      v16bf k1f = frag_b_rowmajor(Kb, 3 * HH, j0 + 16, dk * 32, LL, lane);
      s0 = WMMA_BF16(qa[dk], k0f, s0);
      s1 = WMMA_BF16(qa[dk], k1f, s1);
    }
    // Online softmax update (row stats replicated across the 16-lane half).
#pragma unroll
    for (int r = 0; r < 8; r++) {
      int qi = qt * 16 + r + 8 * half;
      float v0 = s0[r] * scale, v1 = s1[r] * scale;
      if (causal) {
        if (j0 + lc > qi) v0 = -1.0e30f;
        if (j0 + 16 + lc > qi) v1 = -1.0e30f;
      }
      float mx = fmaxf(v0, v1);
#pragma unroll
      for (int d = 1; d < 16; d <<= 1) mx = fmaxf(mx, __shfl_xor(mx, d, 32));
      float nm = fmaxf(rm[r], mx);
      float fs = __expf(rm[r] - nm);
      rm[r] = nm;
      float p0 = __expf(v0 - nm), p1 = __expf(v1 - nm);
      float sum = p0 + p1;
#pragma unroll
      for (int d = 1; d < 16; d <<= 1) sum += __shfl_xor(sum, d, 32);
      rs[r] = rs[r] * fs + sum;
#pragma unroll
      for (int dt = 0; dt < 6; dt++) o[dt][r] *= fs;
      Pld[(r + 8 * half) * 32 + lc] = p0;
      Pld[(r + 8 * half) * 32 + 16 + lc] = p1;
    }
    __syncthreads();  // single-wave WG: lowers to dscnt wait
    {
      const float* prow = Pld + lc * 32;
      v16bf pa;
#pragma unroll
      for (int i = 0; i < 8; i++) {
        pa[i] = (__bf16)prow[half * 16 + i];
        pa[8 + i] = (__bf16)prow[half * 16 + 8 + i];
      }
#pragma unroll
      for (int dt = 0; dt < 6; dt++) {
        int dcol = dt * 16 + lc;
        const bf16_t* vp = Vb + (size_t)(j0 + half * 16) * (3 * HH) + dcol;
        v16bf vb;
#pragma unroll
        for (int i = 0; i < 16; i++) vb[i] = vp[(size_t)i * (3 * HH)];
        o[dt] = WMMA_BF16(pa, vb, o[dt]);
      }
    }
    __syncthreads();
  }
#pragma unroll
  for (int r = 0; r < 8; r++) {
    int qi = qt * 16 + r + 8 * half;
    float inv = 1.0f / rs[r];
#pragma unroll
    for (int dt = 0; dt < 6; dt++) {
      ctx[((size_t)b * LL + qi) * HH + h * HD + dt * 16 + lc] = (__bf16)(o[dt][r] * inv);
    }
  }
}

// combined = alpha*attn + (1-alpha)*lstm  (fp32 + bf16 copies)
__global__ void k_combine(const float* __restrict__ attn, const float* __restrict__ lstm,
                          const float* __restrict__ res_gate, float* __restrict__ comb,
                          bf16_t* __restrict__ comb_bf) {
  float alpha = sigm(res_gate[0]);
  size_t n = (size_t)BB * LL * HH;
  for (size_t i = blockIdx.x * (size_t)blockDim.x + threadIdx.x; i < n;
       i += (size_t)gridDim.x * blockDim.x) {
    float v = alpha * attn[i] + (1.0f - alpha) * lstm[i];
    comb[i] = v;
    comb_bf[i] = (__bf16)v;
  }
}

// Per-row: LayerNorm(h1) -> exact GELU -> gate = sigmoid(dot(.,Wg2)+bg2);
// gated = combined * gate (bf16 for the final GEMM). One 192-thread WG/row.
__global__ __launch_bounds__(192) void k_gate(const float* __restrict__ h1,
                                              const float* __restrict__ ln_g,
                                              const float* __restrict__ ln_b,
                                              const float* __restrict__ Wg2,
                                              const float* __restrict__ bg2,
                                              const float* __restrict__ comb,
                                              bf16_t* __restrict__ gated_bf) {
  __shared__ float red[GHH];
  int row = blockIdx.x, tid = threadIdx.x;
  float x = h1[(size_t)row * GHH + tid];

  red[tid] = x;
  __syncthreads();
  for (int s = GHH / 2; s > 0; s >>= 1) {
    if (tid < s) red[tid] += red[tid + s];
    __syncthreads();
  }
  float mu = red[0] / (float)GHH;
  __syncthreads();

  float dx = x - mu;
  red[tid] = dx * dx;
  __syncthreads();
  for (int s = GHH / 2; s > 0; s >>= 1) {
    if (tid < s) red[tid] += red[tid + s];
    __syncthreads();
  }
  float var = red[0] / (float)GHH;
  __syncthreads();

  float hn = dx * rsqrtf(var + 1e-5f) * ln_g[tid] + ln_b[tid];
  float g = 0.5f * hn * (1.0f + erff(hn * 0.7071067811865475f));  // exact GELU

  red[tid] = g * Wg2[tid];
  __syncthreads();
  for (int s = GHH / 2; s > 0; s >>= 1) {
    if (tid < s) red[tid] += red[tid + s];
    __syncthreads();
  }
  float gate = sigm(red[0] + bg2[0]);

  const float* crow = comb + (size_t)row * HH;
  bf16_t* grow = gated_bf + (size_t)row * HH;
  grow[tid] = (__bf16)(crow[tid] * gate);
  grow[tid + GHH] = (__bf16)(crow[tid + GHH] * gate);
}

// ---------------------------------------------------------------------------
// Host launcher
// ---------------------------------------------------------------------------
extern "C" void kernel_launch(void* const* d_in, const int* in_sizes, int n_in,
                              void* d_out, int out_size, void* d_ws, size_t ws_size,
                              hipStream_t stream) {
  (void)in_sizes; (void)n_in; (void)out_size; (void)ws_size;
  const int*   cap   = (const int*)d_in[0];
  const float* vis   = (const float*)d_in[1];
  const float* emb   = (const float*)d_in[2];
  const float* Wih   = (const float*)d_in[3];
  const float* Whh   = (const float*)d_in[4];
  const float* bih   = (const float*)d_in[5];
  const float* bhh   = (const float*)d_in[6];
  const float* Wqkv  = (const float*)d_in[7];
  const float* bqkv  = (const float*)d_in[8];
  const float* lsc   = (const float*)d_in[9];
  const float* gsc   = (const float*)d_in[10];
  const float* Wout  = (const float*)d_in[11];
  const float* bout  = (const float*)d_in[12];
  const float* Wg1   = (const float*)d_in[13];
  const float* bg1   = (const float*)d_in[14];
  const float* ln_g  = (const float*)d_in[15];
  const float* ln_b  = (const float*)d_in[16];
  const float* Wg2   = (const float*)d_in[17];
  const float* bg2   = (const float*)d_in[18];
  const float* rgate = (const float*)d_in[19];
  const float* Wproj = (const float*)d_in[20];
  const float* bproj = (const float*)d_in[21];

  const int M = BB * LL;  // 16384

  char* ws = (char*)d_ws;
  size_t cur = 0;
  auto alloc = [&](size_t bytes) -> char* {
    char* p = ws + cur;
    cur += (bytes + 255) & ~(size_t)255;
    return p;
  };
  bf16_t* x_bf     = (bf16_t*)alloc((size_t)M * XX * 2);
  bf16_t* Wih_bf   = (bf16_t*)alloc((size_t)G4 * XX * 2);
  bf16_t* Whh_bf   = (bf16_t*)alloc((size_t)G4 * HH * 2);
  bf16_t* Wqkv_bf  = (bf16_t*)alloc((size_t)3 * HH * HH * 2);
  bf16_t* Wout_bf  = (bf16_t*)alloc((size_t)HH * HH * 2);
  bf16_t* Wg1_bf   = (bf16_t*)alloc((size_t)GHH * HH * 2);
  bf16_t* Wproj_bf = (bf16_t*)alloc((size_t)VV * HH * 2);
  float*  pre      = (float*)alloc((size_t)M * G4 * 4);
  float*  lstm_f   = (float*)alloc((size_t)M * HH * 4);
  bf16_t* lstm_bf  = (bf16_t*)alloc((size_t)M * HH * 2);
  bf16_t* qkv_bf   = (bf16_t*)alloc((size_t)M * 3 * HH * 2);
  bf16_t* ctx_bf   = (bf16_t*)alloc((size_t)M * HH * 2);
  float*  attn_f   = (float*)alloc((size_t)M * HH * 4);
  float*  comb_f   = (float*)alloc((size_t)M * HH * 4);
  bf16_t* comb_bf  = (bf16_t*)alloc((size_t)M * HH * 2);
  float*  h1_f     = (float*)alloc((size_t)M * GHH * 4);
  bf16_t* gated_bf = (bf16_t*)alloc((size_t)M * HH * 2);

  auto cvt = [&](const float* s, bf16_t* d, size_t n) {
    int blocks = (int)((n + 2047) / 2048);
    if (blocks > 4096) blocks = 4096;
    if (blocks < 1) blocks = 1;
    k_f32_to_bf16<<<blocks, 256, 0, stream>>>(s, d, n);
  };
  cvt(Wih,   Wih_bf,   (size_t)G4 * XX);
  cvt(Whh,   Whh_bf,   (size_t)G4 * HH);
  cvt(Wqkv,  Wqkv_bf,  (size_t)3 * HH * HH);
  cvt(Wout,  Wout_bf,  (size_t)HH * HH);
  cvt(Wg1,   Wg1_bf,   (size_t)GHH * HH);
  cvt(Wproj, Wproj_bf, (size_t)VV * HH);

  k_build_x<<<4096, 256, 0, stream>>>(cap, emb, vis, x_bf);

  auto gemm_f = [&](const bf16_t* A, const bf16_t* W, const float* b1, const float* b2,
                    float* C, int N, int K) {
    dim3 g((N + 255) / 256, M / 16);
    k_gemm<false><<<g, 256, 0, stream>>>(A, W, b1, b2, (void*)C, M, N, K);
  };
  auto gemm_b = [&](const bf16_t* A, const bf16_t* W, const float* b1, const float* b2,
                    bf16_t* C, int N, int K) {
    dim3 g((N + 255) / 256, M / 16);
    k_gemm<true><<<g, 256, 0, stream>>>(A, W, b1, b2, (void*)C, M, N, K);
  };

  // pre = x @ Wih^T + bih + bhh        (16384 x 1536, K=576)
  gemm_f(x_bf, Wih_bf, bih, bhh, pre, G4, XX);

  // Sequential LSTM over 512 steps (one WGP, h in LDS, c in registers).
  k_lstm<<<1, 768, 0, stream>>>(pre, Whh_bf, vis, lstm_f, lstm_bf);

  // qkv = lstm_out @ Wqkv^T + bqkv    (16384 x 1152, K=384) -> bf16
  gemm_b(lstm_bf, Wqkv_bf, bqkv, nullptr, qkv_bf, 3 * HH, HH);

  // Fused attention: grid (q-tiles, heads, batch), one wave per WG.
  k_attn<<<dim3(LL / 16, NHD, BB), 32, 0, stream>>>(qkv_bf, lsc, gsc, ctx_bf);

  // attn_out = ctx @ Wout^T + bout    (16384 x 384)
  gemm_f(ctx_bf, Wout_bf, bout, nullptr, attn_f, HH, HH);

  // combined = alpha*attn + (1-alpha)*lstm
  k_combine<<<4096, 256, 0, stream>>>(attn_f, lstm_f, rgate, comb_f, comb_bf);

  // h1 = combined @ Wg1^T + bg1       (16384 x 192)
  gemm_f(comb_bf, Wg1_bf, bg1, nullptr, h1_f, GHH, HH);

  // LayerNorm + GELU + sigmoid gate -> gated_bf = combined * gate
  k_gate<<<M, GHH, 0, stream>>>(h1_f, ln_g, ln_b, Wg2, bg2, comb_f, gated_bf);

  // logits = gated @ Wproj^T + bproj  (16384 x 5000) -> fp32 d_out
  gemm_f(gated_bf, Wproj_bf, bproj, nullptr, (float*)d_out, VV, HH);
}
